// BatchLSTMCRFTagger_80719615361542
// MI455X (gfx1250) — matrix operand order repaired and checked by reading, not compiled, not run on hardware
//
#include <hip/hip_runtime.h>

// ---------------------------------------------------------------------------
// BiLSTM-CRF tagger for MI455X (gfx1250, wave32, WMMA bf16)
// V=50000, E=256, H=512, L=2, T=256, B=64, K=32
// ---------------------------------------------------------------------------

#define T_  256
#define B_  64
#define E_  256
#define H_  512
#define K_  32
#define G4H_ 2048   // 4*H

typedef __attribute__((ext_vector_type(16))) __bf16 v16bf;
typedef __attribute__((ext_vector_type(8)))  __bf16 v8bf;
typedef __attribute__((ext_vector_type(8)))  float  v8f;

__device__ __forceinline__ float sigm(float x) { return 1.0f / (1.0f + __expf(-x)); }

// One wave accumulates a 16x16 f32 tile: C += A(16xKd) * B^T where
// Aln = &A[row(lane), 0] (row-major, contiguous in k)
// Bln = &W[col(lane), 0] (W is (N, Kd) row-major, i.e. B column contiguous in k)
// A and B both use contiguous k order, k = 16*lh + s (16-bit A 16x32; B 32x16).
__device__ __forceinline__ v8f gemm_k(const __bf16* __restrict__ Aln,
                                      const __bf16* __restrict__ Bln,
                                      int Kd, int lh, v8f acc) {
  for (int k0 = 0; k0 < Kd; k0 += 32) {
    v8bf a0 = *(const v8bf*)(Aln + k0 + lh * 16);
    v8bf a1 = *(const v8bf*)(Aln + k0 + lh * 16 + 8);
    v16bf a = __builtin_shufflevector(a0, a1, 0,1,2,3,4,5,6,7,8,9,10,11,12,13,14,15);
    v16bf b = *(const v16bf*)(Bln + k0 + lh * 16);
    acc = __builtin_amdgcn_wmma_f32_16x16x32_bf16(false, a, false, b,
                                                  (short)0, acc, false, false);
  }
  return acc;
}

// ---------------------------------------------------------------------------
// Prep kernels
// ---------------------------------------------------------------------------
__global__ void cvt_bf16_kernel(const float* __restrict__ in, __bf16* __restrict__ out, int n) {
  for (int i = blockIdx.x * blockDim.x + threadIdx.x; i < n; i += gridDim.x * blockDim.x)
    out[i] = (__bf16)in[i];
}

__global__ void bias_add_kernel(const float* __restrict__ a, const float* __restrict__ b,
                                float* __restrict__ out, int n) {
  int i = blockIdx.x * blockDim.x + threadIdx.x;
  if (i < n) out[i] = a[i] + b[i];
}

__global__ void zero_kernel(unsigned int* __restrict__ p, int nwords) {
  for (int i = blockIdx.x * blockDim.x + threadIdx.x; i < nwords; i += gridDim.x * blockDim.x)
    p[i] = 0u;
}

// Embedding gather -> bf16. grid = T*B blocks, block = E threads.
__global__ void embed_kernel(const int* __restrict__ sent, const float* __restrict__ emb,
                             __bf16* __restrict__ x0) {
  size_t row = blockIdx.x;
  int e = threadIdx.x;
  int v = sent[row];
  x0[row * E_ + e] = (__bf16)emb[(size_t)v * E_ + e];
}

// ---------------------------------------------------------------------------
// One bidirectional LSTM timestep (both directions in one launch).
// grid = 64 blocks (dir = blk>>5, nTile = blk&31), block = 512 threads (16 waves).
// Each block: all 64 batch rows x one 16-wide H column slice x all 4 gates.
//   wave w: gate = w&3, rowTile = w>>2  -> one 16x16 WMMA tile.
// g = x_t @ W_ih^T + h_prev @ W_hh^T  (bias added in gating phase)
// h double-buffered across steps to avoid cross-block races; c partitioned.
// ---------------------------------------------------------------------------
__global__ __launch_bounds__(512)
void lstm_step_kernel(const __bf16* __restrict__ Xin, int Din,
                      __bf16* __restrict__ Xout,
                      const __bf16* __restrict__ WihF, const __bf16* __restrict__ WihR,
                      const __bf16* __restrict__ WhhF, const __bf16* __restrict__ WhhR,
                      const float* __restrict__ biasF, const float* __restrict__ biasR,
                      const __bf16* __restrict__ hPrevF, __bf16* __restrict__ hNextF,
                      const __bf16* __restrict__ hPrevR, __bf16* __restrict__ hNextR,
                      float* __restrict__ cF, float* __restrict__ cR,
                      int t) {
  __shared__ float lds[4][B_][16];

  const int dir = blockIdx.x >> 5;
  const int nT  = blockIdx.x & 31;
  const int tt  = dir ? (T_ - 1 - t) : t;

  const __bf16* Wih   = dir ? WihR   : WihF;
  const __bf16* Whh   = dir ? WhhR   : WhhF;
  const float*  bias  = dir ? biasR  : biasF;
  const __bf16* hPrev = dir ? hPrevR : hPrevF;
  __bf16*       hNext = dir ? hNextR : hNextF;
  float*        cSt   = dir ? cR     : cF;

  const int tid  = threadIdx.x;
  const int wave = tid >> 5;
  const int lane = tid & 31;
  const int gate = wave & 3;
  const int rowT = wave >> 2;
  const int lh   = lane >> 4;
  const int l16  = lane & 15;

  const int rowG = rowT * 16 + l16;              // batch row handled by this lane
  const int colW = gate * H_ + nT * 16 + l16;    // output column in 4H gate space

  v8f acc = {};
  // input projection slice
  acc = gemm_k(Xin + ((size_t)tt * B_ + rowG) * (size_t)Din,
               Wih + (size_t)colW * (size_t)Din, Din, lh, acc);
  // recurrent projection slice
  acc = gemm_k(hPrev + (size_t)rowG * H_,
               Whh + (size_t)colW * H_, H_, lh, acc);

  // scatter D tile to LDS (rows r + 8*lh, col lane&15)
#pragma unroll
  for (int r = 0; r < 8; ++r)
    lds[gate][rowT * 16 + r + lh * 8][l16] = acc[r];
  __syncthreads();

  // gate fusion: 64 rows x 16 cols = 1024 elems, 2 per thread
#pragma unroll
  for (int j = 0; j < 2; ++j) {
    int idx = tid + j * 512;
    int row = idx >> 4;          // batch
    int col = idx & 15;
    int nH  = nT * 16 + col;     // position within H
    float gi = lds[0][row][col] + bias[nH];
    float gf = lds[1][row][col] + bias[H_ + nH];
    float gg = lds[2][row][col] + bias[2 * H_ + nH];
    float go = lds[3][row][col] + bias[3 * H_ + nH];
    float c_old = cSt[row * H_ + nH];
    float cn = sigm(gf) * c_old + sigm(gi) * tanhf(gg);
    float hv = sigm(go) * tanhf(cn);
    cSt[row * H_ + nH] = cn;
    __bf16 hb = (__bf16)hv;
    hNext[row * H_ + nH] = hb;
    Xout[((size_t)tt * B_ + row) * (2 * H_) + (size_t)dir * H_ + nH] = hb;
  }
}

// ---------------------------------------------------------------------------
// feats = X2(16384 x 1024) @ W_out^T(1024 x 32) + b_out  -> fp32
// wave per 16x16 tile: 1024 row tiles x 2 col tiles = 2048 waves = 256 blocks
// ---------------------------------------------------------------------------
__global__ __launch_bounds__(256)
void feats_kernel(const __bf16* __restrict__ X2, const __bf16* __restrict__ WoutB,
                  const float* __restrict__ bout, float* __restrict__ feats) {
  int wv   = blockIdx.x * 8 + (threadIdx.x >> 5);
  int colT = wv & 1;
  int rowTile = wv >> 1;
  int lane = threadIdx.x & 31;
  int lh = lane >> 4, l16 = lane & 15;

  v8f acc = {};
  acc = gemm_k(X2 + ((size_t)rowTile * 16 + l16) * (2 * H_),
               WoutB + (size_t)(colT * 16 + l16) * (2 * H_), 2 * H_, lh, acc);
#pragma unroll
  for (int r = 0; r < 8; ++r) {
    int row = rowTile * 16 + r + lh * 8;
    int col = colT * 16 + l16;
    feats[(size_t)row * K_ + col] = acc[r] + bout[col];
  }
}

// ---------------------------------------------------------------------------
// Viterbi decode: single block, dp in LDS, strict > argmax (first-max ties,
// matches jnp.argmax). out = [scores(B)] ++ [paths(T*B) as float].
// ---------------------------------------------------------------------------
__global__ __launch_bounds__(1024)
void viterbi_kernel(const float* __restrict__ feats, const float* __restrict__ trans,
                    const float* __restrict__ startt, const float* __restrict__ stopt,
                    int* __restrict__ bps, float* __restrict__ out) {
  __shared__ float dp[2][B_][K_];
  const int tid = threadIdx.x;

  for (int idx = tid; idx < B_ * K_; idx += 1024) {
    int b = idx >> 5, k = idx & 31;
    dp[0][b][k] = startt[k] + feats[(size_t)b * K_ + k];
  }
  __syncthreads();

  int cur = 0;
  for (int t = 1; t < T_; ++t) {
    for (int idx = tid; idx < B_ * K_; idx += 1024) {
      int b = idx >> 5, j = idx & 31;
      float best = trans[j * K_] + dp[cur][b][0];
      int arg = 0;
#pragma unroll
      for (int k = 1; k < K_; ++k) {
        float v = trans[j * K_ + k] + dp[cur][b][k];
        if (v > best) { best = v; arg = k; }
      }
      dp[cur ^ 1][b][j] = best + feats[((size_t)t * B_ + b) * K_ + j];
      bps[((size_t)(t - 1) * B_ + b) * K_ + j] = arg;
    }
    __syncthreads();
    cur ^= 1;
  }

  if (tid < B_) {
    int b = tid;
    float best = dp[cur][b][0] + stopt[0];
    int arg = 0;
    for (int k = 1; k < K_; ++k) {
      float v = dp[cur][b][k] + stopt[k];
      if (v > best) { best = v; arg = k; }
    }
    out[b] = best;
    out[B_ + (size_t)(T_ - 1) * B_ + b] = (float)arg;
    int tag = arg;
    for (int t = T_ - 2; t >= 0; --t) {
      tag = bps[((size_t)t * B_ + b) * K_ + tag];
      out[B_ + (size_t)t * B_ + b] = (float)tag;
    }
  }
}

// ---------------------------------------------------------------------------
// Host orchestration
// ---------------------------------------------------------------------------
extern "C" void kernel_launch(void* const* d_in, const int* in_sizes, int n_in,
                              void* d_out, int out_size, void* d_ws, size_t ws_size,
                              hipStream_t stream) {
  (void)in_sizes; (void)n_in; (void)out_size; (void)ws_size;

  const int*   sent = (const int*)d_in[0];
  const float* emb  = (const float*)d_in[1];
  const float* wih[2][2] = {{(const float*)d_in[2],  (const float*)d_in[6]},
                            {(const float*)d_in[10], (const float*)d_in[14]}};
  const float* whh[2][2] = {{(const float*)d_in[3],  (const float*)d_in[7]},
                            {(const float*)d_in[11], (const float*)d_in[15]}};
  const float* bih[2][2] = {{(const float*)d_in[4],  (const float*)d_in[8]},
                            {(const float*)d_in[12], (const float*)d_in[16]}};
  const float* bhh[2][2] = {{(const float*)d_in[5],  (const float*)d_in[9]},
                            {(const float*)d_in[13], (const float*)d_in[17]}};
  const float* Wout  = (const float*)d_in[18];
  const float* bout  = (const float*)d_in[19];
  const float* trans = (const float*)d_in[20];
  const float* start_t = (const float*)d_in[21];
  const float* stop_t  = (const float*)d_in[22];

  // ---- workspace carve-out ----
  char* ws = (char*)d_ws;
  size_t off = 0;
  auto alloc = [&](size_t bytes) -> char* {
    char* p = ws + off;
    off += (bytes + 255) & ~(size_t)255;
    return p;
  };
  const int Din[2] = {E_, 2 * H_};

  __bf16* X0 = (__bf16*)alloc((size_t)T_ * B_ * E_ * 2);
  __bf16* X1 = (__bf16*)alloc((size_t)T_ * B_ * (2 * H_) * 2);
  __bf16* X2 = (__bf16*)alloc((size_t)T_ * B_ * (2 * H_) * 2);
  __bf16* WihB[2][2], *WhhB[2][2];
  float*  biasC[2][2];
  for (int l = 0; l < 2; ++l)
    for (int d = 0; d < 2; ++d) {
      WihB[l][d]  = (__bf16*)alloc((size_t)G4H_ * Din[l] * 2);
      WhhB[l][d]  = (__bf16*)alloc((size_t)G4H_ * H_ * 2);
      biasC[l][d] = (float*)alloc((size_t)G4H_ * 4);
    }
  __bf16* WoutB = (__bf16*)alloc((size_t)K_ * (2 * H_) * 2);
  // h double buffers per dir + c per dir
  __bf16* hBuf = (__bf16*)alloc((size_t)2 * 2 * B_ * H_ * 2); // [dir][pingpong][B*H]
  float*  cBuf = (float*) alloc((size_t)2 * B_ * H_ * 4);     // [dir][B*H]
  float*  feats = (float*)alloc((size_t)T_ * B_ * K_ * 4);
  int*    bps   = (int*)  alloc((size_t)(T_ - 1) * B_ * K_ * 4);

  // ---- weight conversion / bias fusion (once per call, deterministic) ----
  for (int l = 0; l < 2; ++l)
    for (int d = 0; d < 2; ++d) {
      int nih = G4H_ * Din[l];
      cvt_bf16_kernel<<<512, 256, 0, stream>>>(wih[l][d], WihB[l][d], nih);
      cvt_bf16_kernel<<<512, 256, 0, stream>>>(whh[l][d], WhhB[l][d], G4H_ * H_);
      bias_add_kernel<<<(G4H_ + 255) / 256, 256, 0, stream>>>(bih[l][d], bhh[l][d],
                                                              biasC[l][d], G4H_);
    }
  cvt_bf16_kernel<<<64, 256, 0, stream>>>(Wout, WoutB, K_ * (2 * H_));

  // ---- embedding gather ----
  embed_kernel<<<T_ * B_, E_, 0, stream>>>(sent, emb, X0);

  // ---- two bidirectional LSTM layers ----
  const size_t hStride = (size_t)B_ * H_;
  for (int l = 0; l < 2; ++l) {
    zero_kernel<<<256, 256, 0, stream>>>((unsigned int*)hBuf,
                                         (int)(2 * 2 * hStride * 2 / 4));
    zero_kernel<<<256, 256, 0, stream>>>((unsigned int*)cBuf, (int)(2 * hStride));
    const __bf16* Xin  = (l == 0) ? X0 : X1;
    __bf16*       Xout = (l == 0) ? X1 : X2;
    for (int t = 0; t < T_; ++t) {
      int p = t & 1;
      __bf16* hPrevF = hBuf + (size_t)(0 * 2 + p) * hStride;
      __bf16* hNextF = hBuf + (size_t)(0 * 2 + (p ^ 1)) * hStride;
      __bf16* hPrevR = hBuf + (size_t)(1 * 2 + p) * hStride;
      __bf16* hNextR = hBuf + (size_t)(1 * 2 + (p ^ 1)) * hStride;
      lstm_step_kernel<<<64, 512, 0, stream>>>(
          Xin, Din[l], Xout,
          WihB[l][0], WihB[l][1], WhhB[l][0], WhhB[l][1],
          biasC[l][0], biasC[l][1],
          hPrevF, hNextF, hPrevR, hNextR,
          cBuf + 0 * hStride, cBuf + 1 * hStride, t);
    }
  }

  // ---- output projection ----
  feats_kernel<<<256, 256, 0, stream>>>(X2, WoutB, bout, feats);

  // ---- Viterbi decode -> d_out ----
  viterbi_kernel<<<1, 1024, 0, stream>>>(feats, trans, start_t, stop_t, bps,
                                         (float*)d_out);
}
